// RNN_35424890257580
// MI455X (gfx1250) — matrix operand order repaired and checked by reading, not compiled, not run on hardware
//
#include <hip/hip_runtime.h>
#include <hip/hip_bf16.h>
#include <cstdint>
#include <cstddef>

// ---------------- types ----------------
typedef __bf16 bf16_t;
typedef bf16_t v16bf __attribute__((ext_vector_type(16)));
typedef bf16_t v8bf  __attribute__((ext_vector_type(8)));
typedef float  v8f   __attribute__((ext_vector_type(8)));

#define T_STEPS 4096
#define I_DIM   1024
#define H_DIM   1024
#define O_DIM   50257
#define SCAN_BLOCKS 128
#define J_PER_BLK   8
#define ROWS_PER_BLK (3 * J_PER_BLK)

// LDS tile pitch: 64 bf16 row (128B) + 16B TDM pad = 144B = 72 elems.
// 144B = 36 banks stride; 36 = 4*9, 9 coprime to 16 => the 16 fragment rows
// hit 16 disjoint 4-bank spans => conflict-free ds_load_b128.
#define TILE_PITCH 72

// ---------------- TDM availability ----------------
#if defined(__HIP_DEVICE_COMPILE__) && \
    __has_builtin(__builtin_amdgcn_tensor_load_to_lds) && \
    __has_builtin(__builtin_amdgcn_s_wait_tensorcnt)
#define USE_TDM 1
#else
#define USE_TDM 0
#endif

#if USE_TDM
typedef unsigned int u32x4 __attribute__((ext_vector_type(4)));
typedef int          i32x4 __attribute__((ext_vector_type(4)));
typedef int          i32x8 __attribute__((ext_vector_type(8)));

// Issue one 2D bf16 tile load (tile_rows x tile_k) global->LDS via the
// Tensor Data Mover. Descriptor bitfields per CDNA5 ISA 8.3/8.4.
// LDS destination gets a 16B pad after every 128B row (pitch 144B).
__device__ __forceinline__ void tdm_load_tile_bf16(
    unsigned lds_byte_off, const bf16_t* gsrc,
    unsigned tile_k, unsigned tile_rows,
    unsigned rem_k, unsigned rem_rows, unsigned row_stride_elems) {
  unsigned long long ga = (unsigned long long)(uintptr_t)gsrc;
  u32x4 g0;
  g0[0] = 1u;                                   // count=1, user descriptor
  g0[1] = lds_byte_off;                         // lds_addr (bytes)
  g0[2] = (unsigned)ga;                         // global_addr[31:0]
  g0[3] = (unsigned)((ga >> 32) & 0x01FFFFFFu)  // global_addr[56:32]
          | (2u << 30);                         // type = 2 ("image")
  i32x8 g1;
  // [17:16] data_size=1 (2B) | [20] pad_enable | [24:22] pad_interval=4
  // (2^(4+1)=32 DWORDs = 128B) | [31:25] pad_amount=3 (4 DWORDs = 16B)
  g1[0] = (int)((1u << 16) | (1u << 20) | (4u << 22) | (3u << 25));
  g1[1] = (int)((rem_k & 0xFFFFu) << 16);                      // tensor_dim0 lo
  g1[2] = (int)(((rem_k >> 16) & 0xFFFFu) |
                ((rem_rows & 0xFFFFu) << 16));                 // dim0 hi | dim1 lo
  g1[3] = (int)(((rem_rows >> 16) & 0xFFFFu) |
                ((tile_k & 0xFFFFu) << 16));                   // dim1 hi | tile_dim0
  g1[4] = (int)(tile_rows & 0xFFFFu);                          // tile_dim1, tile_dim2=0
  g1[5] = (int)row_stride_elems;                               // tensor_dim0_stride lo32
  g1[6] = 0;
  g1[7] = 0;
  i32x4 z4 = {0, 0, 0, 0};
#if __clang_major__ >= 23
  i32x8 z8 = {0, 0, 0, 0, 0, 0, 0, 0};
  __builtin_amdgcn_tensor_load_to_lds(g0, g1, z4, z4, z8, 0);
#else
  __builtin_amdgcn_tensor_load_to_lds(g0, g1, z4, z4, 0);
#endif
}
#endif  // USE_TDM

// ---------------- conversion / gather kernels ----------------
__global__ __launch_bounds__(256) void gather_cast_kernel(
    const float* __restrict__ embed, const int* __restrict__ tokens,
    bf16_t* __restrict__ x, int Tn, int Idim) {
  size_t idx = (size_t)blockIdx.x * blockDim.x + threadIdx.x;
  size_t total = (size_t)Tn * Idim;
  if (idx >= total) return;
  int t = (int)(idx / Idim);
  int i = (int)(idx % Idim);
  x[idx] = (bf16_t)embed[(size_t)tokens[t] * Idim + i];
}

__global__ __launch_bounds__(256) void cast_f32_bf16_kernel(
    const float* __restrict__ src, bf16_t* __restrict__ dst, size_t n) {
  size_t idx = (size_t)blockIdx.x * blockDim.x + threadIdx.x;
  if (idx < n) dst[idx] = (bf16_t)src[idx];
}

// ---------------- TDM-staged WMMA GEMM ----------------
// Out[M,N] = A[M,K] * B[N,K]^T + bias[N].
// Block = 64x64 output tile, 8 waves (4 m-tiles x 2 n-strips of 32),
// K staged in 64-wide double-buffered LDS tiles fed by the Tensor Data
// Mover (fallback: cooperative vector loads). Per stage: 4 WMMAs/wave.
template <bool OUT_BF16>
__global__ __launch_bounds__(256) void wmma_gemm_kernel(
    const bf16_t* __restrict__ A, const bf16_t* __restrict__ B,
    const float* __restrict__ bias, void* __restrict__ Out,
    int M, int N, int K) {
  __shared__ bf16_t sA[2][64 * TILE_PITCH];   // 2 x 9 KB
  __shared__ bf16_t sB[2][64 * TILE_PITCH];   // 2 x 9 KB

  const int nb  = N >> 6;
  const int bm  = blockIdx.x / nb;
  const int bn  = blockIdx.x % nb;
  const int tid = threadIdx.x;
  const int lane = tid & 31;
  const int wv   = tid >> 5;
  const int wm   = wv & 3;       // m-tile within block (0..3)
  const int wn   = wv >> 2;      // n-strip within block (0..1)
  const int row  = lane & 15;
  const int hi   = lane >> 4;

  const int row_a = bm << 6;     // block's first A row (M dim)
  const int row_b = bn << 6;     // block's first B row (N dim)
  const int NS = K >> 6;         // number of 64-wide K stages

  v8f acc0 = {}, acc1 = {};

  auto load_stage = [&](int buf, int ks) {
    const int k0 = ks << 6;
#if USE_TDM
    if (wv == 0) {   // one wave drives the TDM; EXEC is ignored by tensor ops
      tdm_load_tile_bf16((unsigned)(uintptr_t)&sA[buf][0],
                         A + (size_t)row_a * K + k0, 64u, 64u,
                         (unsigned)(K - k0), (unsigned)(M - row_a),
                         (unsigned)K);
      tdm_load_tile_bf16((unsigned)(uintptr_t)&sB[buf][0],
                         B + (size_t)row_b * K + k0, 64u, 64u,
                         (unsigned)(K - k0), (unsigned)(N - row_b),
                         (unsigned)K);
    }
#else
    const int r = tid >> 2;
    const int c = (tid & 3) << 4;
    *(v16bf*)&sA[buf][r * TILE_PITCH + c] =
        *(const v16bf*)(A + (size_t)(row_a + r) * K + k0 + c);
    *(v16bf*)&sB[buf][r * TILE_PITCH + c] =
        *(const v16bf*)(B + (size_t)(row_b + r) * K + k0 + c);
#endif
  };

  auto wait_stage = [&]() {
#if USE_TDM
    if (wv == 0) __builtin_amdgcn_s_wait_tensorcnt(0);
#endif
    __syncthreads();
  };

  load_stage(0, 0);
  wait_stage();
  int buf = 0;
  for (int ks = 0; ks < NS; ++ks) {
    if (ks + 1 < NS) load_stage(buf ^ 1, ks + 1);  // prefetch next stage

    // Fragment reads from LDS, contiguous k order k = kk + 16*hi + s for A and B (wave32):
    const bf16_t* sa  = &sA[buf][(wm * 16 + row) * TILE_PITCH + hi * 16];
    const bf16_t* sb0 = &sB[buf][(wn * 32 + row) * TILE_PITCH + hi * 16];
    const bf16_t* sb1 = sb0 + 16 * TILE_PITCH;
#pragma unroll
    for (int kk = 0; kk < 64; kk += 32) {
      v8bf alo = *(const v8bf*)(sa + kk);        // K = kk+hi*16 .. +7
      v8bf ahi = *(const v8bf*)(sa + kk + 8);    // K = kk+hi*16+8 .. +15
      v16bf a = __builtin_shufflevector(alo, ahi,
          0, 1, 2, 3, 4, 5, 6, 7, 8, 9, 10, 11, 12, 13, 14, 15);
      v16bf b0 = *(const v16bf*)(sb0 + kk);
      v16bf b1 = *(const v16bf*)(sb1 + kk);
      acc0 = __builtin_amdgcn_wmma_f32_16x16x32_bf16(
          false, a, false, b0, (short)0, acc0, false, false);
      acc1 = __builtin_amdgcn_wmma_f32_16x16x32_bf16(
          false, a, false, b1, (short)0, acc1, false, false);
    }
    wait_stage();
    buf ^= 1;
  }

  // Epilogue: C/D vgpr v -> row (hi*8 + v), col (lane&15)
  const int m0    = row_a + wm * 16;
  const int n0    = row_b + wn * 32;
  const int col0  = n0 + row;
  const int rbase = m0 + (hi << 3);
  const float bs0 = bias[col0];
  const float bs1 = bias[col0 + 16];
  if (OUT_BF16) {
    bf16_t* o = (bf16_t*)Out;
#pragma unroll
    for (int v = 0; v < 8; ++v) {
      o[(size_t)(rbase + v) * N + col0]      = (bf16_t)(acc0[v] + bs0);
      o[(size_t)(rbase + v) * N + col0 + 16] = (bf16_t)(acc1[v] + bs1);
    }
  } else {
    float* o = (float*)Out;
#pragma unroll
    for (int v = 0; v < 8; ++v) {
      o[(size_t)(rbase + v) * N + col0]      = acc0[v] + bs0;
      o[(size_t)(rbase + v) * N + col0 + 16] = acc1[v] + bs1;
    }
  }
}

// ---------------- scan init ----------------
__global__ __launch_bounds__(256) void scan_init_kernel(
    float* __restrict__ hbuf, unsigned* __restrict__ sync) {
  int t = threadIdx.x;
  for (int i = t; i < 2 * H_DIM; i += 256) hbuf[i] = 0.0f;
  if (t == 0) { sync[0] = 0u; sync[1] = 0u; }
}

// ---------------- persistent GRU scan ----------------
__global__ __launch_bounds__(256) void gru_scan_kernel(
    const float* __restrict__ w_hh, const float* __restrict__ b_hh,
    const float* __restrict__ gi, float* __restrict__ hbuf,
    unsigned* __restrict__ sync, int nblocks, int Tn) {
  __shared__ bf16_t sw[ROWS_PER_BLK * H_DIM];  // 48 KB
  __shared__ float  sh_h[H_DIM];               // 4 KB
  __shared__ float  gh_s[ROWS_PER_BLK];
  __shared__ float  bhh_s[ROWS_PER_BLK];

  const int tid   = threadIdx.x;
  const int lane  = tid & 31;
  const int wv    = tid >> 5;
  const int gbase = blockIdx.x * J_PER_BLK;

  for (int e = tid; e < ROWS_PER_BLK * H_DIM; e += 256) {
    int rr = e >> 10, k = e & (H_DIM - 1);
    int gate = rr >> 3, jj = rr & 7;
    int rowg = gate * H_DIM + gbase + jj;
    sw[e] = (bf16_t)w_hh[(size_t)rowg * H_DIM + k];
  }
  if (tid < ROWS_PER_BLK) {
    int gate = tid >> 3, jj = tid & 7;
    bhh_s[tid] = b_hh[gate * H_DIM + gbase + jj];
  }
  __syncthreads();

  unsigned* cnt = sync;
  unsigned* gen = sync + 1;

  for (int s = 0; s < Tn; ++s) {
    const float* hin  = hbuf + ((s) & 1) * H_DIM;
    float*       hout = hbuf + ((s + 1) & 1) * H_DIM;

    for (int k = tid; k < H_DIM; k += 256) sh_h[k] = hin[k];
    __syncthreads();

#pragma unroll
    for (int q = 0; q < 3; ++q) {
      int rr = wv * 3 + q;
      const bf16_t* wr = &sw[rr << 10];
      float sum = 0.0f;
#pragma unroll 8
      for (int k = lane; k < H_DIM; k += 32)
        sum += (float)wr[k] * sh_h[k];
#pragma unroll
      for (int off = 16; off > 0; off >>= 1)
        sum += __shfl_down(sum, off, 32);
      if (lane == 0) gh_s[rr] = sum + bhh_s[rr];
    }
    __syncthreads();

    if (tid < J_PER_BLK) {
      int jg = gbase + tid;
      const float* g = gi + (size_t)s * (3 * H_DIM);
      float ir  = g[jg];
      float iz  = g[H_DIM + jg];
      float inn = g[2 * H_DIM + jg];
      float r = 1.0f / (1.0f + __expf(-(ir + gh_s[tid])));
      float z = 1.0f / (1.0f + __expf(-(iz + gh_s[J_PER_BLK + tid])));
      float n = tanhf(inn + r * gh_s[2 * J_PER_BLK + tid]);
      hout[jg] = (1.0f - z) * n + z * sh_h[jg];
    }

    __syncthreads();
    if (tid == 0) {
      __threadfence();
      unsigned g0 = __hip_atomic_load(gen, __ATOMIC_RELAXED,
                                      __HIP_MEMORY_SCOPE_AGENT);
      unsigned prev = __hip_atomic_fetch_add(cnt, 1u, __ATOMIC_ACQ_REL,
                                             __HIP_MEMORY_SCOPE_AGENT);
      if (prev == (unsigned)nblocks - 1u) {
        __hip_atomic_store(cnt, 0u, __ATOMIC_RELAXED,
                           __HIP_MEMORY_SCOPE_AGENT);
        __hip_atomic_fetch_add(gen, 1u, __ATOMIC_RELEASE,
                               __HIP_MEMORY_SCOPE_AGENT);
      } else {
        while (__hip_atomic_load(gen, __ATOMIC_ACQUIRE,
                                 __HIP_MEMORY_SCOPE_AGENT) == g0)
          __builtin_amdgcn_s_sleep(1);
      }
    }
    __syncthreads();
  }
}

// ---------------- output head ----------------
__global__ __launch_bounds__(256) void logits_kernel(
    const float* __restrict__ w, const float* __restrict__ b,
    const float* __restrict__ h, float* __restrict__ out, int O) {
  int o = blockIdx.x * 8 + (threadIdx.x >> 5);
  int lane = threadIdx.x & 31;
  if (o >= O) return;
  const float* wr = w + (size_t)o * H_DIM;
  float sum = 0.0f;
#pragma unroll 8
  for (int k = lane; k < H_DIM; k += 32) sum += wr[k] * h[k];
#pragma unroll
  for (int off = 16; off > 0; off >>= 1) sum += __shfl_down(sum, off, 32);
  if (lane == 0) out[o] = sum + b[o];
}

// ---------------- log-softmax ----------------
__global__ __launch_bounds__(256) void lse_kernel(
    const float* __restrict__ x, float* __restrict__ red, int n) {
  __shared__ float sd[256];
  int tid = threadIdx.x;
  float m = -3.402823466e38f;
  for (int i = tid; i < n; i += 256) m = fmaxf(m, x[i]);
  sd[tid] = m;
  __syncthreads();
  for (int s = 128; s > 0; s >>= 1) {
    if (tid < s) sd[tid] = fmaxf(sd[tid], sd[tid + s]);
    __syncthreads();
  }
  float mx = sd[0];
  __syncthreads();
  float sum = 0.0f;
  for (int i = tid; i < n; i += 256) sum += __expf(x[i] - mx);
  sd[tid] = sum;
  __syncthreads();
  for (int s = 128; s > 0; s >>= 1) {
    if (tid < s) sd[tid] += sd[tid + s];
    __syncthreads();
  }
  if (tid == 0) red[0] = mx + logf(sd[0]);
}

__global__ __launch_bounds__(256) void lsm_write_kernel(
    const float* __restrict__ x, const float* __restrict__ red,
    float* __restrict__ out, int n) {
  int i = blockIdx.x * blockDim.x + threadIdx.x;
  if (i < n) out[i] = x[i] - red[0];
}

// ---------------- launcher ----------------
static inline size_t align256(size_t x) { return (x + 255) & ~(size_t)255; }

extern "C" void kernel_launch(void* const* d_in, const int* in_sizes, int n_in,
                              void* d_out, int out_size, void* d_ws, size_t ws_size,
                              hipStream_t stream) {
  (void)in_sizes; (void)n_in; (void)out_size; (void)ws_size;
  const int*   tokens = (const int*)d_in[0];
  const float* embed  = (const float*)d_in[1];
  const float* i2h_w  = (const float*)d_in[2];
  const float* i2h_b  = (const float*)d_in[3];
  const float* w_ih   = (const float*)d_in[4];
  const float* w_hh   = (const float*)d_in[5];
  const float* b_ih   = (const float*)d_in[6];
  const float* b_hh   = (const float*)d_in[7];
  const float* h2o_w  = (const float*)d_in[8];
  const float* h2o_b  = (const float*)d_in[9];
  float* out = (float*)d_out;

  uint8_t* ws = (uint8_t*)d_ws;
  size_t off = 0;
  bf16_t* x_bf   = (bf16_t*)(ws + off); off = align256(off + (size_t)T_STEPS * I_DIM * 2);
  bf16_t* w1_bf  = (bf16_t*)(ws + off); off = align256(off + (size_t)H_DIM * I_DIM * 2);
  bf16_t* xin_bf = (bf16_t*)(ws + off); off = align256(off + (size_t)T_STEPS * H_DIM * 2);
  bf16_t* w2_bf  = (bf16_t*)(ws + off); off = align256(off + (size_t)3 * H_DIM * H_DIM * 2);
  float*  gi     = (float*)(ws + off);  off = align256(off + (size_t)T_STEPS * 3 * H_DIM * 4);
  float*  hbuf   = (float*)(ws + off);  off = align256(off + (size_t)2 * H_DIM * 4);
  float*  logits = (float*)(ws + off);  off = align256(off + (size_t)O_DIM * 4);
  float*  red    = (float*)(ws + off);  off = align256(off + 4);
  unsigned* sync = (unsigned*)(ws + off); off = align256(off + 8);

  // 1) gather embeddings -> bf16, cast weights -> bf16
  {
    size_t n = (size_t)T_STEPS * I_DIM;
    gather_cast_kernel<<<(unsigned)((n + 255) / 256), 256, 0, stream>>>(
        embed, tokens, x_bf, T_STEPS, I_DIM);
  }
  {
    size_t n = (size_t)H_DIM * I_DIM;
    cast_f32_bf16_kernel<<<(unsigned)((n + 255) / 256), 256, 0, stream>>>(
        i2h_w, w1_bf, n);
  }
  {
    size_t n = (size_t)3 * H_DIM * H_DIM;
    cast_f32_bf16_kernel<<<(unsigned)((n + 255) / 256), 256, 0, stream>>>(
        w_ih, w2_bf, n);
  }

  // 2) xin = x @ i2h_w^T + i2h_b   (bf16 out)   M=4096 N=1024 K=1024
  {
    int M = T_STEPS, N = H_DIM, K = I_DIM;
    int blocks = (M >> 6) * (N >> 6);
    wmma_gemm_kernel<true><<<blocks, 256, 0, stream>>>(
        x_bf, w1_bf, i2h_b, (void*)xin_bf, M, N, K);
  }
  // 3) gi = xin @ w_ih^T + b_ih    (f32 out)    M=4096 N=3072 K=1024
  {
    int M = T_STEPS, N = 3 * H_DIM, K = H_DIM;
    int blocks = (M >> 6) * (N >> 6);
    wmma_gemm_kernel<false><<<blocks, 256, 0, stream>>>(
        xin_bf, w2_bf, b_ih, (void*)gi, M, N, K);
  }

  // 4) sequential GRU scan (persistent kernel, w_hh pinned in LDS)
  scan_init_kernel<<<1, 256, 0, stream>>>(hbuf, sync);
  gru_scan_kernel<<<SCAN_BLOCKS, 256, 0, stream>>>(
      w_hh, b_hh, gi, hbuf, sync, SCAN_BLOCKS, T_STEPS);

  // 5) output head + log-softmax (h_final in hbuf[0] since T is even)
  logits_kernel<<<(O_DIM + 7) / 8, 256, 0, stream>>>(
      h2o_w, h2o_b, hbuf, logits, O_DIM);
  lse_kernel<<<1, 256, 0, stream>>>(logits, red, O_DIM);
  lsm_write_kernel<<<(O_DIM + 255) / 256, 256, 0, stream>>>(
      logits, red, out, O_DIM);
}
